// Class1ProcessingModel_5274219839710
// MI455X (gfx1250) — matrix operand order repaired and checked by reading, not compiled, not run on hardware
//
#include <hip/hip_runtime.h>
#include <hip/hip_bf16.h>

typedef __attribute__((ext_vector_type(16))) _Float16 v16h;
typedef __attribute__((ext_vector_type(8)))  float    v8f;

#define SEQ_L   35
#define CIN     21
#define NFLANK  10
#define CFLANK  10
#define FEAT    128
#define KW      9
#define HID     64
#define KCONV   189          // KW*CIN
#define KPAD    192          // padded to 6 chunks of 32
#define LPAD    48           // 35 positions padded to 3 M-tiles

// LDS row strides padded for 64-bank conflict-free fragment loads
#define SAPAD   200          // im2col row stride (f16): 100 dwords -> 16 distinct banks
#define CPAD    136          // conv row stride (f16):    68 dwords -> 16 distinct banks
#define HPAD    72           // hidden row stride (f16):  36 dwords -> 16 distinct banks

// workspace layout in uint32 units (f16 pairs)
#define CONV_PACK_OFF   0
#define CONV_PACK_UINTS (8*6*32*8)     // 8 ntiles * 6 kchunks * 32 lanes * 8 dwords = 12288
#define N_PACK_OFF      (CONV_PACK_OFF + CONV_PACK_UINTS)
#define N_PACK_UINTS    (4*4*32*8)     // 4096
#define C_PACK_OFF      (N_PACK_OFF + N_PACK_UINTS)
#define C_PACK_UINTS    (4*4*32*8)     // 4096
#define WS_TOTAL_UINTS  (C_PACK_OFF + C_PACK_UINTS)

__device__ __forceinline__ unsigned pack2h(float a, float b) {
    union { _Float16 h[2]; unsigned u; } x;
    x.h[0] = (_Float16)a; x.h[1] = (_Float16)b;
    return x.u;
}

// ---------------- weight pre-pack kernel --------------------------------
// Packs f32 weights into the per-lane WMMA B-fragment layout (f16 pairs):
// fragment (ntile, kchunk) = 32 lanes x 8 dwords contiguous.
// Lane n = lane&15 selects column, lane>>4 selects K half; dword j holds
// K = kchunk*32 + (lane>>4)*16 + 2j and +1.
__global__ void prep_pack(const float* __restrict__ conv_w,
                          const float* __restrict__ n_w1,
                          const float* __restrict__ c_w1,
                          unsigned* __restrict__ ws) {
    int t = blockIdx.x * blockDim.x + threadIdx.x;
    if (t < CONV_PACK_UINTS) {
        int j = t & 7, lane = (t >> 3) & 31, frag = t >> 8;
        int kc = frag % 6, nt = frag / 6;
        int kk = kc * 32 + ((lane >> 4) * 16) + 2 * j;
        int n  = nt * 16 + (lane & 15);
        // im2col conv weight: W[kk][f] = conv_w[(k*CIN+c)*F + f] = conv_w[kk*F + f]
        float a = (kk     < KCONV) ? conv_w[kk * FEAT + n]       : 0.f;
        float b = (kk + 1 < KCONV) ? conv_w[(kk + 1) * FEAT + n] : 0.f;
        ws[CONV_PACK_OFF + t] = pack2h(a, b);
    } else if (t < N_PACK_OFF + N_PACK_UINTS) {
        int u = t - N_PACK_OFF;
        int j = u & 7, lane = (u >> 3) & 31, frag = u >> 8;
        int kc = frag & 3, nt = frag >> 2;
        int kk = kc * 32 + ((lane >> 4) * 16) + 2 * j;
        int n  = nt * 16 + (lane & 15);
        ws[N_PACK_OFF + u] = pack2h(n_w1[kk * HID + n], n_w1[(kk + 1) * HID + n]);
    } else if (t < WS_TOTAL_UINTS) {
        int u = t - C_PACK_OFF;
        int j = u & 7, lane = (u >> 3) & 31, frag = u >> 8;
        int kc = frag & 3, nt = frag >> 2;
        int kk = kc * 32 + ((lane >> 4) * 16) + 2 * j;
        int n  = nt * 16 + (lane & 15);
        ws[C_PACK_OFF + u] = pack2h(c_w1[kk * HID + n], c_w1[(kk + 1) * HID + n]);
    }
}

// load a pre-packed B fragment (2 x global_load_b128)
__device__ __forceinline__ v16h load_bfrag(const unsigned* __restrict__ base, int lane) {
    union { v16h h; uint4 u[2]; } tmp;
    const uint4* p = (const uint4*)(base + lane * 8);
    tmp.u[0] = p[0];
    tmp.u[1] = p[1];
    return tmp.h;
}

// load a 16x32 f16 A fragment from LDS (2 x ds_load_b128 per lane)
// srcRow points at element [rowA][kchunk*32] of a padded-f16 LDS matrix.
__device__ __forceinline__ v16h load_afrag_lds(const _Float16* srcRow, int khalf) {
    union { v16h h; unsigned u[8]; } af;
    const unsigned* baseU = (const unsigned*)srcRow;
#pragma unroll
    for (int i = 0; i < 4; ++i) af.u[i] = baseU[(khalf * 16 + 2 * i) >> 1];
#pragma unroll
    for (int i = 4; i < 8; ++i) af.u[i] = baseU[(khalf * 16 + 2 * i) >> 1];
    return af.h;
}

// ---------------- shared MLP stage (hidden GEMM + tanh head) ------------
__device__ void postnet(const unsigned* __restrict__ packBase,
                        const float* __restrict__ b1,
                        const float* __restrict__ w2, float b2v,
                        const _Float16* __restrict__ sC,
                        _Float16* __restrict__ sHid,
                        float* __restrict__ sY,
                        int wave, int lane) {
    int mloc = lane & 15, khalf = lane >> 4;
    int nt = wave;  // 4 waves cover HID=64 (4 n-tiles)
    v16h Bf[4];
#pragma unroll
    for (int kc = 0; kc < 4; ++kc)
        Bf[kc] = load_bfrag(packBase + (nt * 4 + kc) * 256, lane);
    float bias = b1[nt * 16 + mloc];
    for (int mt = 0; mt < 3; ++mt) {
        v8f acc = {};
        int rowA = mt * 16 + mloc;
#pragma unroll
        for (int kc = 0; kc < 4; ++kc) {
            v16h a = load_afrag_lds(&sC[rowA * CPAD + kc * 32], khalf);
            acc = __builtin_amdgcn_wmma_f32_16x16x32_f16(false, a, false, Bf[kc],
                                                         (short)0, acc, false, false);
        }
#pragma unroll
        for (int r = 0; r < 8; ++r) {
            int m = khalf * 8 + r;
            float v = acc[r] + bias;
            v = v > 0.f ? v : 0.f;
            sHid[(mt * 16 + m) * HPAD + nt * 16 + mloc] = (_Float16)v;
        }
    }
    __syncthreads();
    // 64 -> 1 head: one thread per padded position
    int tid = wave * 32 + lane;
    if (tid < LPAD) {
        float acc2 = b2v;
#pragma unroll 8
        for (int h = 0; h < HID; ++h) acc2 += (float)sHid[tid * HPAD + h] * w2[h];
        sY[tid] = tanhf(acc2);
    }
    __syncthreads();
}

// ---------------- fused per-batch-row kernel ----------------------------
__global__ __launch_bounds__(128)
void fused_model(const float* __restrict__ seq, const int* __restrict__ plen_p,
                 const unsigned* __restrict__ ws,
                 const float* __restrict__ conv_b,
                 const float* __restrict__ n_b1, const float* __restrict__ n_w2,
                 const float* __restrict__ n_b2,
                 const float* __restrict__ c_b1, const float* __restrict__ c_w2,
                 const float* __restrict__ c_b2,
                 const float* __restrict__ navg_w, const float* __restrict__ navg_b,
                 const float* __restrict__ cavg_w, const float* __restrict__ cavg_b,
                 const float* __restrict__ out_w, const float* __restrict__ out_b,
                 float* __restrict__ out, int B) {
    __shared__ __align__(16) float    sSeq[SEQ_L * CIN];     //  2.9 KB
    __shared__ __align__(16) _Float16 sA[LPAD * SAPAD];      // 18.8 KB im2col (f16, padded)
    __shared__ __align__(16) _Float16 sC[LPAD * CPAD];       // 12.8 KB conv relu (f16, padded)
    __shared__ __align__(16) _Float16 sHid[LPAD * HPAD];     //  6.8 KB hidden relu (f16, padded)
    __shared__ __align__(16) float    sYn[LPAD], sYc[LPAD];

    int b = blockIdx.x;
    if (b >= B) return;
    int tid  = threadIdx.x;
    int wave = tid >> 5, lane = tid & 31;

    const float* row = seq + (size_t)b * (SEQ_L * CIN);
    __builtin_prefetch(row, 0, 3);  // global_prefetch_b8

    for (int i = tid; i < SEQ_L * CIN; i += 128) sSeq[i] = row[i];
    __syncthreads();

    // build padded im2col A[l][kk], kk = k*CIN + c, input pos = l + k - 4
    for (int i = tid; i < LPAD * KPAD; i += 128) {
        int l = i / KPAD, kk = i - l * KPAD;
        float v = 0.f;
        if (l < SEQ_L && kk < KCONV) {
            int k = kk / CIN, c = kk - k * CIN;
            int p = l + k - (KW / 2);
            if (p >= 0 && p < SEQ_L) v = sSeq[p * CIN + c];
        }
        sA[l * SAPAD + kk] = (_Float16)v;
    }
    __syncthreads();

    // ---- conv GEMM: (48 x 192) x (192 x 128), f16 WMMA, f32 accum ----
    int mloc = lane & 15, khalf = lane >> 4;
    for (int ni = 0; ni < 2; ++ni) {
        int nt = wave + ni * 4;  // 8 n-tiles over FEAT=128
        v16h Bf[6];
#pragma unroll
        for (int kc = 0; kc < 6; ++kc)
            Bf[kc] = load_bfrag(ws + CONV_PACK_OFF + (nt * 6 + kc) * 256, lane);
        float bias = conv_b[nt * 16 + mloc];
        for (int mt = 0; mt < 3; ++mt) {
            v8f acc = {};
            int rowA = mt * 16 + mloc;
#pragma unroll
            for (int kc = 0; kc < 6; ++kc) {
                v16h a = load_afrag_lds(&sA[rowA * SAPAD + kc * 32], khalf);
                acc = __builtin_amdgcn_wmma_f32_16x16x32_f16(false, a, false, Bf[kc],
                                                             (short)0, acc, false, false);
            }
#pragma unroll
            for (int r = 0; r < 8; ++r) {
                int m = khalf * 8 + r;
                float v = acc[r] + bias;
                v = v > 0.f ? v : 0.f;
                sC[(mt * 16 + m) * CPAD + nt * 16 + mloc] = (_Float16)v;
            }
        }
    }
    __syncthreads();

    // ---- N-terminal and C-terminal per-position nets ----
    postnet(ws + N_PACK_OFF, n_b1, n_w2, n_b2[0], sC, sHid, sYn, wave, lane);
    postnet(ws + C_PACK_OFF, c_b1, c_w2, c_b2[0], sC, sHid, sYc, wave, lane);

    // ---- scalar epilogue on wave 0 ----
    if (tid < 32) {
        int plen = plen_p[b];
        float cleaved_n = sYn[NFLANK];
        float mn = 0.f;
        for (int p = NFLANK + 1; p < NFLANK + plen; ++p) mn = fmaxf(mn, sYn[p] + 1.f);
        float maxpool_n = 1.f - mn;
        float cleaved_c = sYc[NFLANK + plen - 1];
        float mc = 0.f;
        for (int p = NFLANK; p < NFLANK + plen - 1; ++p) mc = fmaxf(mc, sYc[p] + 1.f);
        float maxpool_c = 1.f - mc;

        int start = NFLANK + plen;
        int endc  = start + CFLANK < SEQ_L ? start + CFLANK : SEQ_L;
        int cnti  = endc - start; if (cnti < 1) cnti = 1;
        float cnt = (float)cnti;

        float pn = 0.f, pc = 0.f;
        for (int f = lane; f < FEAT; f += 32) {
            float s = 0.f;
            for (int p = 0; p < NFLANK; ++p) s += (float)sC[p * CPAD + f];
            pn += (s * (1.f / NFLANK)) * navg_w[f];
            float s2 = 0.f;
            for (int p = start; p < endc; ++p) s2 += (float)sC[p * CPAD + f] + 1.f;
            pc += (s2 / cnt - 1.f) * cavg_w[f];
        }
#pragma unroll
        for (int off = 16; off > 0; off >>= 1) {   // wave32 butterfly reduce
            pn += __shfl_xor(pn, off, 32);
            pc += __shfl_xor(pc, off, 32);
        }
        if (lane == 0) {
            float avg_n = tanhf(pn + navg_b[0]);
            float avg_c = tanhf(pc + cavg_b[0]);
            float z = cleaved_n * out_w[0] + maxpool_n * out_w[1] + avg_n * out_w[2]
                    + cleaved_c * out_w[3] + maxpool_c * out_w[4] + avg_c * out_w[5]
                    + out_b[0];
            out[b] = 1.f / (1.f + expf(-z));
        }
    }
}

// ---------------- host launcher -----------------------------------------
extern "C" void kernel_launch(void* const* d_in, const int* in_sizes, int n_in,
                              void* d_out, int out_size, void* d_ws, size_t ws_size,
                              hipStream_t stream) {
    const float* seq    = (const float*)d_in[0];
    const int*   plen   = (const int*)  d_in[1];
    const float* conv_w = (const float*)d_in[2];
    const float* conv_b = (const float*)d_in[3];
    const float* n_w1   = (const float*)d_in[4];
    const float* n_b1   = (const float*)d_in[5];
    const float* n_w2   = (const float*)d_in[6];
    const float* n_b2   = (const float*)d_in[7];
    const float* c_w1   = (const float*)d_in[8];
    const float* c_b1   = (const float*)d_in[9];
    const float* c_w2   = (const float*)d_in[10];
    const float* c_b2   = (const float*)d_in[11];
    const float* navg_w = (const float*)d_in[12];
    const float* navg_b = (const float*)d_in[13];
    const float* cavg_w = (const float*)d_in[14];
    const float* cavg_b = (const float*)d_in[15];
    const float* out_w  = (const float*)d_in[16];
    const float* out_b  = (const float*)d_in[17];

    int B = in_sizes[0] / (SEQ_L * CIN);
    unsigned* ws = (unsigned*)d_ws;

    prep_pack<<<(WS_TOTAL_UINTS + 127) / 128, 128, 0, stream>>>(conv_w, n_w1, c_w1, ws);
    fused_model<<<B, 128, 0, stream>>>(seq, plen, ws, conv_b,
                                       n_b1, n_w2, n_b2, c_b1, c_w2, c_b2,
                                       navg_w, navg_b, cavg_w, cavg_b,
                                       out_w, out_b, (float*)d_out, B);
}
